// MultiColumnDeformConvBlock_47390669144783
// MI455X (gfx1250) — matrix operand order repaired and checked by reading, not compiled, not run on hardware
//
#include <hip/hip_runtime.h>
#include <hip/hip_bf16.h>

typedef __attribute__((ext_vector_type(16))) _Float16 v16h;
typedef __attribute__((ext_vector_type(8)))  _Float16 v8h;
typedef __attribute__((ext_vector_type(2)))  _Float16 v2h;
typedef __attribute__((ext_vector_type(8)))  float    v8f;

#define CIN      256
#define HH       64
#define WW       64
#define BB       2
#define OUT_PER  256
#define OC_TOTAL 768
#define NOFF     166
#define TPX      16    // pixels per tile (one 16-wide row segment)
#define SPAD     264   // padded LDS row length in halves (256 + 8 -> bank-conflict-free)

// ---------------------------------------------------------------------------
// Repack weights (O, C, KK) fp32  ->  (O, j*256 + c) f16, contiguous in K.
// Makes the WMMA A-matrix stream a plain contiguous global_load_b128 stream.
// ---------------------------------------------------------------------------
__global__ __launch_bounds__(256)
void repack_w(const float* __restrict__ src, _Float16* __restrict__ dst, int KK)
{
    int total = OUT_PER * CIN * KK;
    int idx = blockIdx.x * blockDim.x + threadIdx.x;   // linear over dst (c fastest)
    if (idx >= total) return;
    int c = idx % CIN;
    int t = idx / CIN;
    int j = t % KK;
    int o = t / KK;
    dst[idx] = (_Float16)src[((size_t)o * CIN + c) * KK + j];
}

// ---------------------------------------------------------------------------
// Fused deformable conv via implicit GEMM on v_wmma_f32_16x16x32_f16.
//   grid.x = BB*HH*(WW/16) pixel tiles, grid.y = branch (k = 3/5/7)
//   block  = 256 threads = 8 wave32; wave w owns out-channels [32w, 32w+32)
// ---------------------------------------------------------------------------
__global__ __launch_bounds__(256)
void deform_wmma(const float* __restrict__ feat,
                 const float* __restrict__ offs,
                 const _Float16* __restrict__ wr3,
                 const _Float16* __restrict__ wr5,
                 const _Float16* __restrict__ wr7,
                 float* __restrict__ out)
{
    // Double-buffered sampled panel S[buf][pixel][channel] (f16), padded rows.
    __shared__ __attribute__((aligned(16))) _Float16 S[2][TPX][SPAD];

    const int tid  = threadIdx.x;
    const int lane = tid & 31;
    const int wave = tid >> 5;

    // ---- branch parameters
    const int br = blockIdx.y;
    int KK, offc0, ochBase;
    const _Float16* WrB;
    if (br == 0)      { KK = 9;  offc0 = 0;  ochBase = 0;   WrB = wr3; }
    else if (br == 1) { KK = 25; offc0 = 18; ochBase = 256; WrB = wr5; }
    else              { KK = 49; offc0 = 68; ochBase = 512; WrB = wr7; }
    const int Ktot = KK * CIN;

    // ---- pixel tile: 16 consecutive x in one image row
    const int t  = blockIdx.x;
    const int b  = t / (HH * (WW / TPX));
    const int rr = t % (HH * (WW / TPX));
    const int y  = rr / (WW / TPX);
    const int x0 = (rr % (WW / TPX)) * TPX;

    // ---- gather assignment: each thread = 1 pixel x 16 channels
    const int p  = tid & 15;
    const int cb = (tid >> 4) * 16;
    const float* fbase = feat + ((size_t)b * CIN + cb) * (HH * WW);
    const float  xpix  = (float)(x0 + p);

    // ---- A-fragment row bases (ISA 16-bit A 16x32 layout: lanes>=16 take K+8)
    const int klo   = (lane & 16) ? 8 : 0;
    const _Float16* arow0 = WrB + (size_t)(wave * 32 +      (lane & 15)) * Ktot + klo;
    const _Float16* arow1 = WrB + (size_t)(wave * 32 + 16 + (lane & 15)) * Ktot + klo;

    v8f acc0 = {}; v8f acc1 = {};

    for (int j = 0; j < KK; ++j) {
        const int bu = j & 1;

        // ===== phase 2: bilinear-sample 16 channels of pixel p for tap j =====
        {
            // base grid cancels in the reference: sample at (y+off_y, x+off_x)
            size_t ob = (((size_t)b * NOFF + offc0 + 2 * j) * HH + y) * WW + (x0 + p);
            float oy = offs[ob];
            float ox = offs[ob + (size_t)HH * WW];
            float ysf = (float)y + oy;
            float xsf = xpix + ox;
            float y0f = floorf(ysf), x0f = floorf(xsf);
            float dy = ysf - y0f,   dx = xsf - x0f;
            int iy0 = (int)y0f, ix0 = (int)x0f;
            int iy1 = iy0 + 1,  ix1 = ix0 + 1;
            float vy0 = (iy0 >= 0 && iy0 < HH) ? 1.f : 0.f;
            float vy1 = (iy1 >= 0 && iy1 < HH) ? 1.f : 0.f;
            float vx0 = (ix0 >= 0 && ix0 < WW) ? 1.f : 0.f;
            float vx1 = (ix1 >= 0 && ix1 < WW) ? 1.f : 0.f;
            int cy0 = iy0 < 0 ? 0 : (iy0 > HH - 1 ? HH - 1 : iy0);
            int cy1 = iy1 < 0 ? 0 : (iy1 > HH - 1 ? HH - 1 : iy1);
            int cx0 = ix0 < 0 ? 0 : (ix0 > WW - 1 ? WW - 1 : ix0);
            int cx1 = ix1 < 0 ? 0 : (ix1 > WW - 1 ? WW - 1 : ix1);
            int i00 = cy0 * WW + cx0, i01 = cy0 * WW + cx1;
            int i10 = cy1 * WW + cx0, i11 = cy1 * WW + cx1;
            float w00 = (1.f - dy) * (1.f - dx) * vy0 * vx0;
            float w01 = (1.f - dy) * dx         * vy0 * vx1;
            float w10 = dy * (1.f - dx)         * vy1 * vx0;
            float w11 = dy * dx                 * vy1 * vx1;
            #pragma unroll
            for (int cc = 0; cc < 16; cc += 2) {
                const float* f0 = fbase + (size_t)(cc + 0) * (HH * WW);
                const float* f1 = fbase + (size_t)(cc + 1) * (HH * WW);
                float s0 = w00 * f0[i00] + w01 * f0[i01] + w10 * f0[i10] + w11 * f0[i11];
                float s1 = w00 * f1[i00] + w01 * f1[i01] + w10 * f1[i10] + w11 * f1[i11];
                v2h pk; pk[0] = (_Float16)s0; pk[1] = (_Float16)s1;
                *(v2h*)&S[bu][p][cb + cc] = pk;   // packed ds_store_b32
            }
        }
        __syncthreads();   // one barrier per tap (double buffering covers WAR)

        // ===== phase 3: 16 WMMAs over K = 256 channels of this tap =====
        const size_t kTap = (size_t)j * CIN;
        if (j + 1 < KK) {  // stream next tap's weight rows toward the WGP
            __builtin_prefetch(arow0 + kTap + CIN, 0, 0);
            __builtin_prefetch(arow1 + kTap + CIN, 0, 0);
        }
        // B-fragment source (ISA 16-bit B 32x16: lane n -> pixel n&15, K+8 for lanes>=16)
        const _Float16* brow = &S[bu][lane & 15][(lane & 16) ? 8 : 0];

        #pragma unroll
        for (int cc = 0; cc < 8; ++cc) {
            const int kb = cc * 32;

            v8h bl = *(const v8h*)(brow + kb);        // ds_load_b128
            v8h bh = *(const v8h*)(brow + kb + 16);   // ds_load_b128
            v16h bf;
            #pragma unroll
            for (int i = 0; i < 8; ++i) { bf[i] = bl[i]; bf[i + 8] = bh[i]; }

            const _Float16* a0p = arow0 + kTap + kb;
            const _Float16* a1p = arow1 + kTap + kb;
            v8h a0l = *(const v8h*)(a0p);             // global_load_b128
            v8h a0h = *(const v8h*)(a0p + 16);
            v8h a1l = *(const v8h*)(a1p);
            v8h a1h = *(const v8h*)(a1p + 16);
            v16h a0, a1;
            #pragma unroll
            for (int i = 0; i < 8; ++i) {
                a0[i] = a0l[i]; a0[i + 8] = a0h[i];
                a1[i] = a1l[i]; a1[i + 8] = a1h[i];
            }

            acc0 = __builtin_amdgcn_wmma_f32_16x16x32_f16(false, a0, false, bf,
                                                          (short)0, acc0, false, false);
            acc1 = __builtin_amdgcn_wmma_f32_16x16x32_f16(false, a1, false, bf,
                                                          (short)0, acc1, false, false);
        }
    }

    // ===== epilogue: D layout -> coalesced 64B row stores =====
    const int px   = x0 + (lane & 15);
    const int mrow = (lane < 16) ? 0 : 8;             // VGPR r holds M=r / M=8+r
    float* o0 = out + (((size_t)b * OC_TOTAL + ochBase + wave * 32 + mrow) * HH + y) * WW + px;
    float* o1 = o0 + (size_t)16 * HH * WW;
    #pragma unroll
    for (int r = 0; r < 8; ++r) o0[(size_t)r * HH * WW] = acc0[r];
    #pragma unroll
    for (int r = 0; r < 8; ++r) o1[(size_t)r * HH * WW] = acc1[r];
}

// ---------------------------------------------------------------------------
extern "C" void kernel_launch(void* const* d_in, const int* in_sizes, int n_in,
                              void* d_out, int out_size, void* d_ws, size_t ws_size,
                              hipStream_t stream)
{
    (void)in_sizes; (void)n_in; (void)out_size; (void)ws_size;
    const float* feat = (const float*)d_in[0];
    const float* offs = (const float*)d_in[1];
    const float* w3   = (const float*)d_in[2];
    const float* w5   = (const float*)d_in[3];
    const float* w7   = (const float*)d_in[4];
    float* out = (float*)d_out;

    // f16 repacked weights live in workspace (~10.4 MB total)
    _Float16* wr3 = (_Float16*)d_ws;
    _Float16* wr5 = wr3 + (size_t)OUT_PER *  9 * CIN;
    _Float16* wr7 = wr5 + (size_t)OUT_PER * 25 * CIN;

    const int n3 = OUT_PER * CIN * 9;
    const int n5 = OUT_PER * CIN * 25;
    const int n7 = OUT_PER * CIN * 49;
    repack_w<<<(n3 + 255) / 256, 256, 0, stream>>>(w3, wr3, 9);
    repack_w<<<(n5 + 255) / 256, 256, 0, stream>>>(w5, wr5, 25);
    repack_w<<<(n7 + 255) / 256, 256, 0, stream>>>(w7, wr7, 49);

    dim3 grid(BB * HH * (WW / TPX), 3);
    deform_wmma<<<grid, 256, 0, stream>>>(feat, offs, wr3, wr5, wr7, out);
}
